// EpiGraphNetSimple_1443109012257
// MI455X (gfx1250) — matrix operand order repaired and checked by reading, not compiled, not run on hardware
//
#include <hip/hip_runtime.h>
#include <hip/hip_bf16.h>
#include <cstdint>

typedef __attribute__((ext_vector_type(16))) _Float16 v16h;
typedef __attribute__((ext_vector_type(8)))  float    v8f;

// ---------------------------------------------------------------------------
// Fast transcendentals: v_exp_f32 + v_rcp_f32 instead of IEEE division chains
// ---------------------------------------------------------------------------
__device__ __forceinline__ float fast_sigmoid(float x) {
  return __builtin_amdgcn_rcpf(1.0f + __expf(-x));
}
__device__ __forceinline__ float fast_tanh(float x) {
  return 1.0f - 2.0f * __builtin_amdgcn_rcpf(__expf(2.0f * x) + 1.0f);
}

// ---------------------------------------------------------------------------
// CDNA5 async global->LDS copy (ASYNCcnt path), GV addressing form
// ---------------------------------------------------------------------------
__device__ __forceinline__ void async_load_b32(uint32_t lds_off, const void* gaddr) {
  asm volatile("global_load_async_to_lds_b32 %0, %1, off"
               :
               : "v"(lds_off), "v"(gaddr)
               : "memory");
}
__device__ __forceinline__ void wait_asynccnt0() {
  asm volatile("s_wait_asynccnt 0" ::: "memory");
}

// ---------------------------------------------------------------------------
// WMMA fragment helpers (16x16x32 f16, contiguous k order: k = 16*(lane/16) + s in A and B)
// ---------------------------------------------------------------------------
__device__ __forceinline__ v16h frag_a_lds(const _Float16* base, int kchunk, int ld) {
  int l = threadIdx.x & 31;
  int m = l & 15;
  int kb = (l & 16) ? 16 : 0;
  const _Float16* row = base + m * ld + kchunk;
  v16h r;
#pragma unroll
  for (int i = 0; i < 8; ++i) r[i] = row[kb + i];
#pragma unroll
  for (int i = 0; i < 8; ++i) r[8 + i] = row[kb + 8 + i];
  return r;
}

// B = W^T where W is [N][K] row-major f32 in global: B(k,n) = W[n][k]
__device__ __forceinline__ v16h frag_b_weights(const float* W, int n0, int kchunk, int ldK) {
  int l = threadIdx.x & 31;
  int n = n0 + (l & 15);
  int kb = (l & 16) ? 16 : 0;
  const float* row = W + (size_t)n * ldK + kchunk + kb;
  v16h r;
#pragma unroll
  for (int i = 0; i < 16; ++i) r[i] = (_Float16)row[i];
  return r;
}

// B = M^T where M is [N][K] row-major f16 in LDS (used for nrm @ nrm^T)
__device__ __forceinline__ v16h frag_bT_lds(const _Float16* base, int n0, int kchunk, int ld) {
  int l = threadIdx.x & 31;
  int n = n0 + (l & 15);
  int kb = (l & 16) ? 16 : 0;
  const _Float16* row = base + n * ld + kchunk + kb;
  v16h r;
#pragma unroll
  for (int i = 0; i < 16; ++i) r[i] = row[i];
  return r;
}

// B = G where G is [Knodes<=16][ld] row-major f16 in LDS; K rows 16..31 zero pad
__device__ __forceinline__ v16h frag_b_nodecols(const _Float16* g16, int n0, int ld) {
  int l = threadIdx.x & 31;
  int n = n0 + (l & 15);
  bool lo = (l & 16) == 0;
  v16h r;
#pragma unroll
  for (int i = 0; i < 16; ++i) r[i] = lo ? g16[i * ld + n] : (_Float16)0.0f;
  return r;
}

__device__ __forceinline__ void store_cd(float* dst, int n0, int ld, v8f c) {
  int l = threadIdx.x & 31;
  int n = n0 + (l & 15);
  int mb = (l & 16) ? 8 : 0;
#pragma unroll
  for (int r = 0; r < 8; ++r) dst[(mb + r) * ld + n] = c[r];
}

__device__ __forceinline__ void store_cd_bias_relu(float* g, _Float16* g16, int n0, int ld,
                                                   v8f c, const float* gb) {
  int l = threadIdx.x & 31;
  int n = n0 + (l & 15);
  int mb = (l & 16) ? 8 : 0;
  float bias = gb[n];
#pragma unroll
  for (int r = 0; r < 8; ++r) {
    float v = fmaxf(c[r] + bias, 0.0f);
    g[(mb + r) * ld + n] = v;
    g16[(mb + r) * ld + n] = (_Float16)v;
  }
}

// ---------------------------------------------------------------------------
// Fused Conv1d(k=5,p=2) + BN(eval, folded) + ReLU + MaxPool(2)
// ---------------------------------------------------------------------------
template <int CIN, int COUT, int LIN>
__global__ void conv_bn_pool_kernel(const float* __restrict__ in, const float* __restrict__ w,
                                    const float* __restrict__ cb, const float* __restrict__ bng,
                                    const float* __restrict__ bnb, const float* __restrict__ bnm,
                                    const float* __restrict__ bnv, float* __restrict__ out) {
  __shared__ float wv[CIN * 5];
  __shared__ float sc[2];
  const int co = blockIdx.y;
  const int b = blockIdx.z;
  const int tid = threadIdx.x;
  if (tid < CIN * 5) wv[tid] = w[(size_t)co * CIN * 5 + tid];
  if (tid == 0) {
    float s = bng[co] * rsqrtf(bnv[co] + 1e-5f);
    sc[0] = s;
    sc[1] = bnb[co] - bnm[co] * s + cb[co] * s;
  }
  __syncthreads();
  const int LOUT = LIN / 2;
  int p = blockIdx.x * blockDim.x + tid;
  if (p >= LOUT) return;
  const float* inb = in + (size_t)b * CIN * LIN;
  const int q = 2 * p;
  float acc0 = 0.0f, acc1 = 0.0f;
#pragma unroll
  for (int ci = 0; ci < CIN; ++ci) {
    const float* row = inb + (size_t)ci * LIN;
    if (ci == 0 && q + 1024 < LIN) __builtin_prefetch(row + q + 1024, 0, 0);
    float v[6];
#pragma unroll
    for (int j = 0; j < 6; ++j) {
      int idx = q - 2 + j;
      v[j] = (idx >= 0 && idx < LIN) ? row[idx] : 0.0f;
    }
#pragma unroll
    for (int j = 0; j < 5; ++j) {
      float wj = wv[ci * 5 + j];
      acc0 += v[j] * wj;
      acc1 += v[j + 1] * wj;
    }
  }
  float s = sc[0], sh = sc[1];
  float y0 = fmaxf(acc0 * s + sh, 0.0f);
  float y1 = fmaxf(acc1 * s + sh, 0.0f);
  out[((size_t)b * COUT + co) * LOUT + p] = fmaxf(y0, y1);
}

// ---------------------------------------------------------------------------
// AdaptiveAvgPool1d(64) + transpose, emitting f16 directly for the LSTM
// ---------------------------------------------------------------------------
__global__ void avgpool_transpose_kernel(const float* __restrict__ in,
                                         _Float16* __restrict__ seq16) {
  int b = blockIdx.x;
  for (int i = threadIdx.x; i < 64 * 64; i += blockDim.x) {
    int t = i >> 6, c = i & 63;
    const float* row = in + ((size_t)b * 64 + c) * 1920 + t * 30;
    float s = 0.0f;
#pragma unroll
    for (int j = 0; j < 30; ++j) s += row[j];
    seq16[((size_t)b * 64 + t) * 64 + c] = (_Float16)(s * (1.0f / 30.0f));
  }
}

// ---------------------------------------------------------------------------
// LSTM layer (H=64, T=64). Block = 16 batch rows x 16 waves; persistent f16
// weight fragments in registers; x_t staged via async global->LDS DMA.
// ---------------------------------------------------------------------------
__global__ void lstm_layer_kernel(const _Float16* __restrict__ X16,  // [B][64][64] f16
                                  const float* __restrict__ wih,     // [256][64]
                                  const float* __restrict__ whh,     // [256][64]
                                  const float* __restrict__ bih, const float* __restrict__ bhh,
                                  float* __restrict__ Y,             // [B][64][64] f32
                                  _Float16* __restrict__ Y16) {      // [B][64][64] f16
  __shared__ _Float16 xt16[16 * 64];
  __shared__ _Float16 h16[16 * 64];
  __shared__ float gates[16 * 256];
  __shared__ float cst[16 * 64];
  __shared__ float bias[256];
  const int tid = threadIdx.x;  // 0..511
  const int wave = tid >> 5;    // 0..15
  const int b0 = blockIdx.x * 16;

  for (int i = tid; i < 16 * 64; i += 512) { h16[i] = (_Float16)0.0f; cst[i] = 0.0f; }
  for (int i = tid; i < 256; i += 512) bias[i] = bih[i] + bhh[i];

  const int n0 = wave * 16;
  v16h fw0 = frag_b_weights(wih, n0, 0, 64);
  v16h fw1 = frag_b_weights(wih, n0, 32, 64);
  v16h fu0 = frag_b_weights(whh, n0, 0, 64);
  v16h fu1 = frag_b_weights(whh, n0, 32, 64);

  // per-thread async-copy source/dest for x_t staging (2 halves = one b32/lane)
  const int j = tid * 2;             // half index 0..1022
  const int xr = j >> 6, xk = j & 63;
  const uint32_t lds_off = (uint32_t)(uintptr_t)&xt16[j];
  const _Float16* gbase = X16 + (((size_t)(b0 + xr)) * 64) * 64 + xk;
  __syncthreads();

  for (int t = 0; t < 64; ++t) {
    async_load_b32(lds_off, gbase + (size_t)t * 64);
    wait_asynccnt0();
    __syncthreads();

    v16h a0 = frag_a_lds(xt16, 0, 64);
    v16h a1 = frag_a_lds(xt16, 32, 64);
    v16h h0 = frag_a_lds(h16, 0, 64);
    v16h h1 = frag_a_lds(h16, 32, 64);
    v8f acc = {};
    acc = __builtin_amdgcn_wmma_f32_16x16x32_f16(false, a0, false, fw0, (short)0, acc, false, false);
    acc = __builtin_amdgcn_wmma_f32_16x16x32_f16(false, a1, false, fw1, (short)0, acc, false, false);
    acc = __builtin_amdgcn_wmma_f32_16x16x32_f16(false, h0, false, fu0, (short)0, acc, false, false);
    acc = __builtin_amdgcn_wmma_f32_16x16x32_f16(false, h1, false, fu1, (short)0, acc, false, false);
    store_cd(gates, n0, 256, acc);
    __syncthreads();

    // elementwise LSTM cell (gate order i,f,g,o) with fast sigmoid/tanh
    for (int i = tid; i < 16 * 64; i += 512) {
      int r = i >> 6, u = i & 63;
      float gi = gates[r * 256 + u] + bias[u];
      float gf = gates[r * 256 + 64 + u] + bias[64 + u];
      float gg = gates[r * 256 + 128 + u] + bias[128 + u];
      float go = gates[r * 256 + 192 + u] + bias[192 + u];
      float c = fast_sigmoid(gf) * cst[i] + fast_sigmoid(gi) * fast_tanh(gg);
      float h = fast_sigmoid(go) * fast_tanh(c);
      cst[i] = c;
      h16[i] = (_Float16)h;
      size_t oidx = (((size_t)(b0 + r)) * 64 + t) * 64 + u;
      Y[oidx] = h;
      Y16[oidx] = (_Float16)h;
    }
    __syncthreads();
  }
}

// ---------------------------------------------------------------------------
// Graph builder + 3-layer GCN + node-max + classifier. One block per batch.
// ---------------------------------------------------------------------------
__device__ __forceinline__ void gcn_layer(float* g, _Float16* g16, float* sbuf, _Float16* s16,
                                          const _Float16* adj16, const float* gw,
                                          const float* gb, int K, int tid, int wave) {
  int ntiles = K >> 4;
  for (int tIdx = wave; tIdx < ntiles; tIdx += 8) {
    v16h a = frag_a_lds(adj16, 0, 32);
    v16h bf = frag_b_nodecols(g16, tIdx * 16, 256);
    v8f acc = {};
    acc = __builtin_amdgcn_wmma_f32_16x16x32_f16(false, a, false, bf, (short)0, acc, false, false);
    store_cd(sbuf, tIdx * 16, 256, acc);
  }
  __syncthreads();
  for (int i = tid; i < 16 * K; i += 256) {
    int m = i / K, f = i - m * K;
    int idx = m * 256 + f;
    s16[idx] = (_Float16)(g[idx] + sbuf[idx]);
  }
  __syncthreads();
  int kc_n = K >> 5;
  for (int nt = wave; nt < 4; nt += 8) {
    v8f acc = {};
    for (int kc = 0; kc < kc_n; ++kc) {
      v16h a = frag_a_lds(s16, kc * 32, 256);
      v16h bf = frag_b_weights(gw, nt * 16, kc * 32, K);
      acc = __builtin_amdgcn_wmma_f32_16x16x32_f16(false, a, false, bf, (short)0, acc, false, false);
    }
    store_cd_bias_relu(g, g16, nt * 16, 256, acc, gb);
  }
  __syncthreads();
}

__global__ void graph_gcn_kernel(const float* __restrict__ Y2,  // [B][64][64]
                                 const float* __restrict__ gw0, const float* __restrict__ gb0,
                                 const float* __restrict__ gw1, const float* __restrict__ gb1,
                                 const float* __restrict__ gw2, const float* __restrict__ gb2,
                                 const float* __restrict__ clw, const float* __restrict__ clb,
                                 float* __restrict__ outp) {
  __shared__ float g[16 * 256];
  __shared__ _Float16 g16[16 * 256];
  __shared__ _Float16 nrm16[16 * 256];
  __shared__ float sbuf[16 * 256];
  __shared__ _Float16 s16[16 * 256];
  __shared__ float Cm[256];
  __shared__ _Float16 adj16[16 * 32];
  __shared__ float stats[32];
  __shared__ float thrs;
  __shared__ float emb[64];
  const int tid = threadIdx.x;  // 256
  const int wave = tid >> 5;
  const int b = blockIdx.x;

  const float* src = Y2 + (size_t)b * 4096;
  for (int i = tid; i < 4096; i += 256) {
    float v = src[i];
    g[i] = v;
    g16[i] = (_Float16)v;
  }
  if (tid == 0) thrs = 0.0f;
  __syncthreads();

  if (tid < 16) {
    float m = 0.0f;
    for (int f = 0; f < 256; ++f) m += g[tid * 256 + f];
    m *= (1.0f / 256.0f);
    float v = 0.0f;
    for (int f = 0; f < 256; ++f) {
      float d = g[tid * 256 + f] - m;
      v += d * d;
    }
    stats[tid * 2] = m;
    stats[tid * 2 + 1] = 1.0f / (sqrtf(v / 255.0f) + 1e-8f);
  }
  __syncthreads();
  for (int i = tid; i < 4096; i += 256) {
    int n = i >> 8;
    nrm16[i] = (_Float16)((g[i] - stats[n * 2]) * stats[n * 2 + 1]);
  }
  __syncthreads();

  if (wave == 0) {
    v8f acc = {};
#pragma unroll
    for (int kc = 0; kc < 8; ++kc) {
      v16h a = frag_a_lds(nrm16, kc * 32, 256);
      v16h bf = frag_bT_lds(nrm16, 0, kc * 32, 256);
      acc = __builtin_amdgcn_wmma_f32_16x16x32_f16(false, a, false, bf, (short)0, acc, false, false);
    }
    int l = tid & 31;
    int n = l & 15;
    int mb = (l & 16) ? 8 : 0;
#pragma unroll
    for (int r = 0; r < 8; ++r) Cm[(mb + r) * 16 + n] = acc[r] * (1.0f / 256.0f);
  }
  __syncthreads();

  {
    float v = Cm[tid];
    int cnt = 0;
    for (int j = 0; j < 256; ++j) {
      float u = Cm[j];
      cnt += (u < v) || (u == v && j < tid);
    }
    if (cnt == 191) thrs = v;
  }
  __syncthreads();

  for (int i = tid; i < 16 * 32; i += 256) {
    int m = i >> 5, k = i & 31;
    float a = (k < 16 && k != m && Cm[m * 16 + k] > thrs) ? 1.0f : 0.0f;
    adj16[i] = (_Float16)a;
  }
  __syncthreads();

  gcn_layer(g, g16, sbuf, s16, adj16, gw0, gb0, 256, tid, wave);
  gcn_layer(g, g16, sbuf, s16, adj16, gw1, gb1, 64, tid, wave);
  gcn_layer(g, g16, sbuf, s16, adj16, gw2, gb2, 64, tid, wave);

  if (tid < 64) {
    float m = g[tid];
    for (int n = 1; n < 16; ++n) m = fmaxf(m, g[n * 256 + tid]);
    emb[tid] = m;
  }
  __syncthreads();

  if (tid < 5) {
    float s = clb[tid];
    for (int f = 0; f < 64; ++f) s += emb[f] * clw[tid * 64 + f];
    outp[(size_t)b * 5 + tid] = s;
  }
}

// ---------------------------------------------------------------------------
// Launch
// ---------------------------------------------------------------------------
extern "C" void kernel_launch(void* const* d_in, const int* in_sizes, int n_in,
                              void* d_out, int out_size, void* d_ws, size_t ws_size,
                              hipStream_t stream) {
  const float* x = (const float*)d_in[0];
  const float* cw0 = (const float*)d_in[1];
  const float* cb0 = (const float*)d_in[2];
  const float* bng0 = (const float*)d_in[3];
  const float* bnb0 = (const float*)d_in[4];
  const float* bnm0 = (const float*)d_in[5];
  const float* bnv0 = (const float*)d_in[6];
  const float* cw1 = (const float*)d_in[7];
  const float* cb1 = (const float*)d_in[8];
  const float* bng1 = (const float*)d_in[9];
  const float* bnb1 = (const float*)d_in[10];
  const float* bnm1 = (const float*)d_in[11];
  const float* bnv1 = (const float*)d_in[12];
  const float* cw2 = (const float*)d_in[13];
  const float* cb2 = (const float*)d_in[14];
  const float* bng2 = (const float*)d_in[15];
  const float* bnb2 = (const float*)d_in[16];
  const float* bnm2 = (const float*)d_in[17];
  const float* bnv2 = (const float*)d_in[18];
  const float* wih0 = (const float*)d_in[19];
  const float* whh0 = (const float*)d_in[20];
  const float* bih0 = (const float*)d_in[21];
  const float* bhh0 = (const float*)d_in[22];
  const float* wih1 = (const float*)d_in[23];
  const float* whh1 = (const float*)d_in[24];
  const float* bih1 = (const float*)d_in[25];
  const float* bhh1 = (const float*)d_in[26];
  const float* gw0 = (const float*)d_in[27];
  const float* gb0 = (const float*)d_in[28];
  const float* gw1 = (const float*)d_in[29];
  const float* gb1 = (const float*)d_in[30];
  const float* gw2 = (const float*)d_in[31];
  const float* gb2 = (const float*)d_in[32];
  const float* clw = (const float*)d_in[33];
  const float* clb = (const float*)d_in[34];

  float* ws = (float*)d_ws;
  float* bufA = ws;                        // 31,457,280 floats
  float* bufB = ws + 31457280u;            // 31,457,280 floats
  float* y1 = ws + 62914560u;              // 1,048,576 floats (f32, layer-1 h seq)
  float* y2 = y1 + 1048576u;               // 1,048,576 floats (f32, layer-2 h seq)
  _Float16* seq16 = (_Float16*)(y2 + 1048576u);  // 1,048,576 halves
  _Float16* y1_16 = seq16 + 1048576u;            // 1,048,576 halves
  _Float16* y2_16 = y1_16 + 1048576u;            // 1,048,576 halves

  conv_bn_pool_kernel<1, 16, 15360><<<dim3(30, 16, 256), 256, 0, stream>>>(
      x, cw0, cb0, bng0, bnb0, bnm0, bnv0, bufA);
  conv_bn_pool_kernel<16, 32, 7680><<<dim3(15, 32, 256), 256, 0, stream>>>(
      bufA, cw1, cb1, bng1, bnb1, bnm1, bnv1, bufB);
  conv_bn_pool_kernel<32, 64, 3840><<<dim3(8, 64, 256), 256, 0, stream>>>(
      bufB, cw2, cb2, bng2, bnb2, bnm2, bnv2, bufA);
  avgpool_transpose_kernel<<<256, 256, 0, stream>>>(bufA, seq16);
  lstm_layer_kernel<<<16, 512, 0, stream>>>(seq16, wih0, whh0, bih0, bhh0, y1, y1_16);
  lstm_layer_kernel<<<16, 512, 0, stream>>>(y1_16, wih1, whh1, bih1, bhh1, y2, y2_16);
  graph_gcn_kernel<<<256, 256, 0, stream>>>(y2, gw0, gb0, gw1, gb1, gw2, gb2, clw, clb,
                                            (float*)d_out);
}
